// SimpleLinearStylizer_15032385536249
// MI455X (gfx1250) — matrix operand order repaired and checked by reading, not compiled, not run on hardware
//
#include <hip/hip_runtime.h>
#include <cstdint>
#include <cstddef>

// ---------------------------------------------------------------------------
// MI455X (gfx1250) implementation. wave32, WMMA f32_16x16x32_f16 everywhere.
// Pipeline (all intermediates in d_ws, L2-resident on a 192MB-L2 part):
//   k_stats(c)  -> mean / rstd per (n,ch)            [reads c: 64MB]
//   k_stats(s)  -> mean / unbiased std per (n,ch)    [reads s: 16MB]
//   k_cvt x8    -> f16 copies of all weights
//   k_content   -> norm(c) -> q,k,v (WMMA) -> per-token softmax -> ct16 (f16)
//   k_style     -> (s-mean) -> 3-layer MLP (WMMA) -> per-block cov partials
//   k_fc        -> reduce cov, fc GEMM (WMMA), M_n = uW @ s_mat (WMMA)
//   k_out       -> out = M_n @ ct (WMMA) * s_std + s_mean*acc_map [writes 64MB]
// ---------------------------------------------------------------------------

#define DEVINL __device__ __forceinline__

typedef __attribute__((ext_vector_type(16))) _Float16 v16h;
typedef __attribute__((ext_vector_type(8)))  _Float16 v8h;
typedef __attribute__((ext_vector_type(8)))  float    v8f;

union V16U { v16h v; v8h h[2]; };

static constexpr int NB = 4, C = 256, S = 16384, SS = 4096, E = 32;

DEVINL v8f wmma_f16(v16h a, v16h b, v8f c) {
  // D = A(16x32 f16) * B(32x16 f16) + C(16x16 f32)
  return __builtin_amdgcn_wmma_f32_16x16x32_f16(false, a, false, b, (short)0, c,
                                                false, false);
}

// A fragment from row-major [M x K] f16 (W[m][k]); tile origin (m0,k0).
// ISA 16-bit A 16x32 layout: lane holds row m0+(l&15); halves i<8 -> k=k0+hi8+i,
// i>=8 -> k=k0+16+hi8+(i-8). Two aligned 16B loads.
DEVINL v16h load_afrag(const _Float16* W, int ldk, int m0, int k0, int lane) {
  int r   = m0 + (lane & 15);
  int hi8 = (lane >> 4) << 3;
  const _Float16* p = W + (size_t)r * ldk + k0 + hi8;
  V16U u;
  u.h[0] = *(const v8h*)(p);
  u.h[1] = *(const v8h*)(p + 16);
  return u.v;
}

// B fragment: source stored as X[n][k] (k contiguous, stride ldk); B(k,n)=X[n][k].
// Lane holds column n0+(l&15); halves i<8 -> k=k0+hi8+i, i>=8 -> k=k0+16+hi8+(i-8) (2x b128).
DEVINL v16h load_bfrag(const _Float16* X, int ldk, int n0, int k0, int lane) {
  int nn   = n0 + (lane & 15);
  int hi8  = (lane >> 4) << 3;
  const _Float16* p = X + (size_t)nn * ldk + k0 + hi8;
  V16U u;
  u.h[0] = *(const v8h*)(p);
  u.h[1] = *(const v8h*)(p + 16);
  return u.v;
}

// Gather A fragment from [K x M] storage: A(m,k) = base[k*stride + m].
DEVINL v16h gather_afrag(const _Float16* base, int stride, int m0, int k0, int lane) {
  int m   = m0 + (lane & 15);
  int hi8 = (lane >> 4) << 3;
  v16h r;
#pragma unroll
  for (int i = 0; i < 16; ++i) {
    int k = k0 + hi8 + ((i < 8) ? i : (i + 8));
    r[i] = base[k * stride + m];
  }
  return r;
}

// Gather B fragment from [K x N] storage: B(k,n) = base[k*stride + n].
DEVINL v16h gather_bfrag(const _Float16* base, int stride, int n0, int k0, int lane) {
  int nn   = n0 + (lane & 15);
  int hi8  = (lane >> 4) << 3;
  v16h r;
#pragma unroll
  for (int i = 0; i < 16; ++i) r[i] = base[(k0 + hi8 + ((i < 8) ? i : (i + 8))) * stride + nn];
  return r;
}

// ---------------------------------------------------------------------------
// Row statistics: mean + (mode 0: rsqrt(biased var + 1e-5), mode 1: unbiased std)
// ---------------------------------------------------------------------------
__global__ void k_stats(const float* __restrict__ x, float* __restrict__ mean_out,
                        float* __restrict__ sec_out, int len, int mode) {
  int row = blockIdx.x;
  const float* p = x + (size_t)row * len;
  float s = 0.f, s2 = 0.f;
  for (int i = threadIdx.x; i < len; i += blockDim.x) {
    float v = p[i];
    s += v; s2 += v * v;
  }
  __shared__ float r1[256], r2[256];
  r1[threadIdx.x] = s; r2[threadIdx.x] = s2;
  __syncthreads();
  for (int o = blockDim.x >> 1; o > 0; o >>= 1) {
    if ((int)threadIdx.x < o) { r1[threadIdx.x] += r1[threadIdx.x + o];
                                r2[threadIdx.x] += r2[threadIdx.x + o]; }
    __syncthreads();
  }
  if (threadIdx.x == 0) {
    float m = r1[0] / (float)len;
    mean_out[row] = m;
    if (mode == 0) {
      float var = fmaxf(r2[0] / (float)len - m * m, 0.f);
      sec_out[row] = rsqrtf(var + 1e-5f);
    } else {
      float varu = fmaxf((r2[0] - (float)len * m * m) / (float)(len - 1), 0.f);
      sec_out[row] = sqrtf(varu);
    }
  }
}

__global__ void k_cvt(const float* __restrict__ src, _Float16* __restrict__ dst, int n) {
  int i = blockIdx.x * blockDim.x + threadIdx.x;
  if (i < n) dst[i] = (_Float16)src[i];
}

// ---------------------------------------------------------------------------
// Content: instance-norm tile of c -> q/k/v via WMMA -> per-token softmax
// attention -> ct16[n][s][e] (f16, e contiguous for later B fragments).
// Block = 128 threads (4 waves), 64 tokens per block; one wave = 16 tokens.
// ---------------------------------------------------------------------------
__global__ void __launch_bounds__(128) k_content(
    const float* __restrict__ c, const float* __restrict__ cmean,
    const float* __restrict__ crstd,
    const _Float16* __restrict__ qW, const _Float16* __restrict__ kW,
    const _Float16* __restrict__ vW,
    const float* __restrict__ qb, const float* __restrict__ kb,
    const float* __restrict__ vb,
    _Float16* __restrict__ ct16) {
  int n = blockIdx.y;
  int tb = blockIdx.x * 64;
  int tid = threadIdx.x, lane = tid & 31, wave = tid >> 5;

  __shared__ _Float16 nct[64][288];          // token-major normalized tile (36.0KB)
  __shared__ float qs[4][16][32], ks[4][16][32], vs[4][16][32];  // 24KB

  for (int idx = tid; idx < C * 64; idx += 128) {
    int ch = idx >> 6, tt = idx & 63;
    float m = cmean[n * C + ch], rs = crstd[n * C + ch];
    float v = c[((size_t)(n * C + ch)) * S + tb + tt];
    nct[tt][ch] = (_Float16)((v - m) * rs);
  }
  __syncthreads();

  int t0 = wave * 16;
  int col = lane & 15, hi8 = (lane >> 4) << 3;

  v8f qa[2] = {}, ka[2] = {}, va[2] = {};
  for (int c0 = 0; c0 < C; c0 += 32) {
    v16h bfr = load_bfrag(&nct[0][0], 288, t0, c0, lane);
#pragma unroll
    for (int mt = 0; mt < 2; ++mt) {
      v16h aq = load_afrag(qW, C, mt * 16, c0, lane);
      v16h ak = load_afrag(kW, C, mt * 16, c0, lane);
      v16h av = load_afrag(vW, C, mt * 16, c0, lane);
      qa[mt] = wmma_f16(aq, bfr, qa[mt]);
      ka[mt] = wmma_f16(ak, bfr, ka[mt]);
      va[mt] = wmma_f16(av, bfr, va[mt]);
    }
  }
#pragma unroll
  for (int mt = 0; mt < 2; ++mt)
#pragma unroll
    for (int r = 0; r < 8; ++r) {
      int e = mt * 16 + r + hi8;
      qs[wave][col][e] = qa[mt][r] + qb[e];
      ks[wave][col][e] = ka[mt][r] + kb[e];
      vs[wave][col][e] = va[mt][r] + vb[e];
    }
  __syncthreads();

  // Per-token attention: lane = output row i (E=32 lanes), softmax over j.
  int i = lane;
  for (int t = 0; t < 16; ++t) {
    float qi = qs[wave][t][i];
    const float* kr = &ks[wave][t][0];
    const float* vr = &vs[wave][t][0];
    float mx = -3.4e38f;
#pragma unroll
    for (int j = 0; j < 32; ++j) mx = fmaxf(mx, qi * kr[j]);
    float sum = 0.f, acc = 0.f;
#pragma unroll
    for (int j = 0; j < 32; ++j) {
      float ee = __expf(qi * kr[j] - mx);
      sum += ee; acc += ee * vr[j];
    }
    int sidx = tb + t0 + t;
    ct16[((size_t)n * S + sidx) * 32 + i] = (_Float16)(acc / sum);
  }
}

// ---------------------------------------------------------------------------
// Style: (s - mean) -> relu(s1) -> relu(s2) -> s3 -> per-block cov partial.
// Block = 128 threads, 64 tokens. covp[n][block][32*32] partials (deterministic).
// ---------------------------------------------------------------------------
__global__ void __launch_bounds__(128) k_style(
    const float* __restrict__ s, const float* __restrict__ smean,
    const _Float16* __restrict__ w1, const float* __restrict__ b1,
    const _Float16* __restrict__ w2, const float* __restrict__ b2,
    const _Float16* __restrict__ w3, const float* __restrict__ b3,
    float* __restrict__ covp) {
  int n = blockIdx.y, tb = blockIdx.x * 64;
  int tid = threadIdx.x, lane = tid & 31, wave = tid >> 5;

  __shared__ union OvlT {
    _Float16 x0[64][288];                               // 36.0KB
    struct { _Float16 h2[64][96]; _Float16 h3[64][32]; } p2;  // 16.0KB
  } ovl;
  __shared__ _Float16 h1[64][160];                      // 20.0KB

  for (int idx = tid; idx < C * 64; idx += 128) {
    int ch = idx >> 6, tt = idx & 63;
    float v = s[((size_t)(n * C + ch)) * SS + tb + tt] - smean[n * C + ch];
    ovl.x0[tt][ch] = (_Float16)v;
  }
  __syncthreads();

  int t0 = wave * 16, col = lane & 15, hi8 = (lane >> 4) << 3;

  // layer1: 128 out, K=256
  for (int mt = 0; mt < 8; ++mt) {
    v8f acc = {};
    for (int c0 = 0; c0 < 256; c0 += 32) {
      v16h a = load_afrag(w1, 256, mt * 16, c0, lane);
      v16h b = load_bfrag(&ovl.x0[0][0], 288, t0, c0, lane);
      acc = wmma_f16(a, b, acc);
    }
#pragma unroll
    for (int r = 0; r < 8; ++r) {
      int o = mt * 16 + r + hi8;
      h1[t0 + col][o] = (_Float16)fmaxf(acc[r] + b1[o], 0.f);
    }
  }
  __syncthreads();

  // layer2: 64 out, K=128 (h2 overlays x0 region; x0 dead after barrier)
  for (int mt = 0; mt < 4; ++mt) {
    v8f acc = {};
    for (int c0 = 0; c0 < 128; c0 += 32) {
      v16h a = load_afrag(w2, 128, mt * 16, c0, lane);
      v16h b = load_bfrag(&h1[0][0], 160, t0, c0, lane);
      acc = wmma_f16(a, b, acc);
    }
#pragma unroll
    for (int r = 0; r < 8; ++r) {
      int o = mt * 16 + r + hi8;
      ovl.p2.h2[t0 + col][o] = (_Float16)fmaxf(acc[r] + b2[o], 0.f);
    }
  }
  __syncthreads();

  // layer3: 32 out, K=64
  for (int mt = 0; mt < 2; ++mt) {
    v8f acc = {};
    for (int c0 = 0; c0 < 64; c0 += 32) {
      v16h a = load_afrag(w3, 64, mt * 16, c0, lane);
      v16h b = load_bfrag(&ovl.p2.h2[0][0], 96, t0, c0, lane);
      acc = wmma_f16(a, b, acc);
    }
#pragma unroll
    for (int r = 0; r < 8; ++r) {
      int o = mt * 16 + r + hi8;
      ovl.p2.h3[t0 + col][o] = (_Float16)(acc[r] + b3[o]);
    }
  }
  __syncthreads();

  // cov partial for this 64-token block: cov += h3 @ h3^T, via WMMA.
  // wave -> one 16x16 output tile of the 32x32 cov; K = 64 tokens (2 steps).
  {
    int mt = wave >> 1, nt = wave & 1;
    v8f acc = {};
    for (int k0 = 0; k0 < 64; k0 += 32) {
      v16h a = gather_afrag(&ovl.p2.h3[0][0], 32, mt * 16, k0, lane);
      v16h b = gather_bfrag(&ovl.p2.h3[0][0], 32, nt * 16, k0, lane);
      acc = wmma_f16(a, b, acc);
    }
#pragma unroll
    for (int r = 0; r < 8; ++r) {
      int e = mt * 16 + r + hi8, f = nt * 16 + col;
      covp[((size_t)n * 64 + blockIdx.x) * 1024 + e * 32 + f] =
          acc[r] * (1.0f / 4096.0f);
    }
  }
}

// ---------------------------------------------------------------------------
// fc: reduce cov partials -> s_fc = cov_flat @ fcW^T + fcb (M=16(4 valid),
// N=1024, K=1024, WMMA) -> s_mat -> M_n = uW @ s_mat_n (WMMA). 1 block.
// ---------------------------------------------------------------------------
__global__ void __launch_bounds__(128) k_fc(
    const float* __restrict__ covp, const _Float16* __restrict__ fcW16,
    const float* __restrict__ fcb, const _Float16* __restrict__ uW16,
    _Float16* __restrict__ Mn16) {
  int tid = threadIdx.x, lane = tid & 31, wave = tid >> 5;
  __shared__ _Float16 covh[16][1024];     // 32KB: padded A rows (rows 4..15 zero)
  __shared__ _Float16 smat[4][32][32];    // 8KB

  for (int idx = tid; idx < 16 * 1024; idx += 128) {
    int rrow = idx >> 10, kcol = idx & 1023;
    float sum = 0.f;
    if (rrow < 4) {
      const float* pp = covp + (size_t)rrow * 64 * 1024 + kcol;
      for (int bb = 0; bb < 64; ++bb) sum += pp[bb * 1024];
    }
    covh[rrow][kcol] = (_Float16)sum;
  }
  __syncthreads();

  int col = lane & 15, hi8 = (lane >> 4) << 3;

  // fc GEMM: wave handles 256 output cols = 16 N-tiles; K=1024 -> 32 steps.
  for (int nt = 0; nt < 16; ++nt) {
    int n0 = wave * 256 + nt * 16;
    v8f acc = {};
    for (int k0 = 0; k0 < 1024; k0 += 32) {
      v16h a = load_afrag(&covh[0][0], 1024, 0, k0, lane);
      v16h b = load_bfrag(fcW16, 1024, n0, k0, lane);  // B(k,o)=fcW[o][k]
      acc = wmma_f16(a, b, acc);
    }
#pragma unroll
    for (int r = 0; r < 8; ++r) {
      int m = r + hi8;                    // padded batch row; valid m<4
      int o = n0 + col;
      if (m < 4) smat[m][o >> 5][o & 31] = (_Float16)(acc[r] + fcb[o]);
    }
  }
  __syncthreads();

  // M_n = uW(256x32) @ s_mat_n(32x32); wave = n; 16 M-tiles x 2 N-tiles, K=32.
  {
    int n = wave;
    v16h b0 = gather_bfrag(&smat[n][0][0], 32, 0, 0, lane);
    v16h b1 = gather_bfrag(&smat[n][0][0], 32, 16, 0, lane);
    for (int mt = 0; mt < 16; ++mt) {
      v16h a = load_afrag(uW16, 32, mt * 16, 0, lane);
#pragma unroll
      for (int ntl = 0; ntl < 2; ++ntl) {
        v8f acc = {};
        acc = wmma_f16(a, ntl ? b1 : b0, acc);
#pragma unroll
        for (int r = 0; r < 8; ++r) {
          int cch = mt * 16 + r + hi8;
          int e = ntl * 16 + col;
          Mn16[((size_t)n * C + cch) * 32 + e] = (_Float16)acc[r];
        }
      }
    }
  }
}

// ---------------------------------------------------------------------------
// Final: out[n][cch][s] = (M_n @ ct)[cch][s] * s_std[n][cch]
//                        + s_mean[n][cch] * acc_map[s]
// Block = 128 threads, 64 tokens; each wave: 16 tokens x 16 M-tiles, K=32.
// ---------------------------------------------------------------------------
__global__ void __launch_bounds__(128) k_out(
    const _Float16* __restrict__ Mn16, const _Float16* __restrict__ ct16,
    const float* __restrict__ sstd, const float* __restrict__ smeanv,
    const float* __restrict__ acc_map, float* __restrict__ out) {
  int n = blockIdx.y, tb = blockIdx.x * 64;
  int tid = threadIdx.x, lane = tid & 31, wave = tid >> 5;
  int col = lane & 15, hi8 = (lane >> 4) << 3;
  int t0 = tb + wave * 16;
  int token = t0 + col;

  v16h b = load_bfrag(ct16 + (size_t)n * S * 32, 32, t0, 0, lane);
  float am = acc_map[token];

  for (int mt = 0; mt < 16; ++mt) {
    v16h a = load_afrag(Mn16 + (size_t)n * C * 32, 32, mt * 16, 0, lane);
    v8f acc = {};
    acc = wmma_f16(a, b, acc);
#pragma unroll
    for (int r = 0; r < 8; ++r) {
      int cch = mt * 16 + r + hi8;
      float sd = sstd[n * C + cch], mn = smeanv[n * C + cch];
      out[((size_t)(n * C + cch)) * S + token] = acc[r] * sd + mn * am;
    }
  }
}

// ---------------------------------------------------------------------------
extern "C" void kernel_launch(void* const* d_in, const int* in_sizes, int n_in,
                              void* d_out, int out_size, void* d_ws, size_t ws_size,
                              hipStream_t stream) {
  (void)in_sizes; (void)n_in; (void)out_size; (void)ws_size;
  const float* c    = (const float*)d_in[0];
  const float* s    = (const float*)d_in[1];
  const float* accm = (const float*)d_in[2];
  const float* qW = (const float*)d_in[3];  const float* qb = (const float*)d_in[4];
  const float* kW = (const float*)d_in[5];  const float* kb = (const float*)d_in[6];
  const float* vW = (const float*)d_in[7];  const float* vb = (const float*)d_in[8];
  const float* uW = (const float*)d_in[9];
  const float* s1W = (const float*)d_in[10]; const float* s1b = (const float*)d_in[11];
  const float* s2W = (const float*)d_in[12]; const float* s2b = (const float*)d_in[13];
  const float* s3W = (const float*)d_in[14]; const float* s3b = (const float*)d_in[15];
  const float* fcW = (const float*)d_in[16]; const float* fcb = (const float*)d_in[17];
  float* out = (float*)d_out;

  char* w = (char*)d_ws;
  auto alloc = [&](size_t bytes) {
    char* p = w;
    w += (bytes + 255) & ~(size_t)255;
    return p;
  };
  float*    cmean = (float*)alloc(NB * C * 4);
  float*    crstd = (float*)alloc(NB * C * 4);
  float*    smean = (float*)alloc(NB * C * 4);
  float*    sstd  = (float*)alloc(NB * C * 4);
  float*    covp  = (float*)alloc((size_t)NB * 64 * 1024 * 4);   // 1MB partials
  _Float16* qW16  = (_Float16*)alloc(E * C * 2);
  _Float16* kW16  = (_Float16*)alloc(E * C * 2);
  _Float16* vW16  = (_Float16*)alloc(E * C * 2);
  _Float16* uW16  = (_Float16*)alloc(C * E * 2);
  _Float16* s1W16 = (_Float16*)alloc(128 * 256 * 2);
  _Float16* s2W16 = (_Float16*)alloc(64 * 128 * 2);
  _Float16* s3W16 = (_Float16*)alloc(32 * 64 * 2);
  _Float16* fcW16 = (_Float16*)alloc((size_t)1024 * 1024 * 2);   // 2MB
  _Float16* ct16  = (_Float16*)alloc((size_t)NB * S * E * 2);    // 4MB
  _Float16* Mn16  = (_Float16*)alloc((size_t)NB * C * E * 2);

  auto cvt = [&](const float* src, _Float16* dst, int n) {
    k_cvt<<<(n + 255) / 256, 256, 0, stream>>>(src, dst, n);
  };
  cvt(qW, qW16, E * C);
  cvt(kW, kW16, E * C);
  cvt(vW, vW16, E * C);
  cvt(uW, uW16, C * E);
  cvt(s1W, s1W16, 128 * 256);
  cvt(s2W, s2W16, 64 * 128);
  cvt(s3W, s3W16, 32 * 64);
  cvt(fcW, fcW16, 1024 * 1024);

  k_stats<<<NB * C, 256, 0, stream>>>(c, cmean, crstd, S, 0);
  k_stats<<<NB * C, 256, 0, stream>>>(s, smean, sstd, SS, 1);

  k_content<<<dim3(S / 64, NB), 128, 0, stream>>>(c, cmean, crstd, qW16, kW16,
                                                  vW16, qb, kb, vb, ct16);
  k_style<<<dim3(SS / 64, NB), 128, 0, stream>>>(s, smean, s1W16, s1b, s2W16,
                                                 s2b, s3W16, s3b, covp);
  k_fc<<<1, 128, 0, stream>>>(covp, fcW16, fcb, uW16, Mn16);
  k_out<<<dim3(S / 64, NB), 128, 0, stream>>>(Mn16, ct16, sstd, smean, accm, out);
}
